// PhaseSyncAttentionV4_11416023072953
// MI455X (gfx1250) — matrix operand order repaired and checked by reading, not compiled, not run on hardware
//
#include <hip/hip_runtime.h>

// ---------------------------------------------------------------------------
// PhaseSyncAttention fused implementation for gfx1250 (MI455X, wave32, WMMA)
// B=4, L=1024, D=256, H=8, S=8, DH=32
// ---------------------------------------------------------------------------

#define B_  4
#define L_  1024
#define D_  256
#define H_  8
#define S_  8
#define DH_ 32
#define BL_ (B_ * L_)        // 4096
#define BH_ (B_ * H_)        // 32

typedef __attribute__((ext_vector_type(16))) __bf16 v16bf;
typedef __attribute__((ext_vector_type(8)))  float  v8f;
typedef unsigned short u16;
typedef unsigned int   u32;

// float -> bf16 (round to nearest even)
static __device__ inline u16 f2bf(float f) {
  u32 u = __builtin_bit_cast(u32, f);
  u += 0x7FFFu + ((u >> 16) & 1u);
  return (u16)(u >> 16);
}

union Frag16 {
  v16bf bf;
  u16   h[16];
  uint4 q[2];
};

// A-matrix fragment, 16x32 bf16, from row-major [16 x >=32], ld in elements.
// Lane m=lane%16 -> row M=m; kh=lane/16 selects K-halves:
//   elems 0..7  -> K = kh*16 + 0..7       (contiguous 16B)
//   elems 8..15 -> K = kh*16 + 8 + 0..7   (contiguous 16B)
static __device__ inline v16bf load_a(const u16* __restrict__ p, int ld) {
  int lane = threadIdx.x & 31;
  int m = lane & 15, kh = lane >> 4;
  const u16* row = p + m * ld;
  Frag16 f;
  f.q[0] = *(const uint4*)(row + kh * 16);
  f.q[1] = *(const uint4*)(row + kh * 16 + 8);
  return f.bf;
}

// B-matrix fragment (32x16, K x N) sourced from a TRANSPOSED row-major
// buffer Mt[N x K]: B[k][n] = Mt[n][k]. Lane n=lane%16 -> column N=n,
// kh=lane/16 -> K = kh*16 + 0..15 (16 contiguous bf16 = 32B per lane).
// Works on both global and LDS-backed pointers (addrspace inferred).
static __device__ inline v16bf load_bt(const u16* p, int ld) {
  int lane = threadIdx.x & 31;
  int n = lane & 15, kh = lane >> 4;
  const u16* row = p + n * ld + kh * 16;
  Frag16 f;
  f.q[0] = *(const uint4*)(row);
  f.q[1] = *(const uint4*)(row + 8);
  return f.bf;
}

static __device__ inline v8f wmma_bf16(v16bf a, v16bf b, v8f c) {
  // (neg_a, A, neg_b, B, c_mod, C, reuse_a, reuse_b)
  return __builtin_amdgcn_wmma_f32_16x16x32_bf16(
      false, a, false, b, (short)0, c, false, false);
}

// 16-lane-group reductions (wave32 halves), valid for C-fragment row groups.
static __device__ inline float rmax16(float v) {
  v = fmaxf(v, __shfl_xor(v, 1, 32));
  v = fmaxf(v, __shfl_xor(v, 2, 32));
  v = fmaxf(v, __shfl_xor(v, 4, 32));
  v = fmaxf(v, __shfl_xor(v, 8, 32));
  return v;
}
static __device__ inline float rsum16(float v) {
  v += __shfl_xor(v, 1, 32);
  v += __shfl_xor(v, 2, 32);
  v += __shfl_xor(v, 4, 32);
  v += __shfl_xor(v, 8, 32);
  return v;
}

// Async global -> LDS copy of 16B per lane (tracked by ASYNCcnt).
static __device__ inline void async_cp16(u32 lds_off, const u16* g) {
  asm volatile("global_load_async_to_lds_b128 %0, %1, off"
               :: "v"(lds_off), "v"(g) : "memory");
}
static __device__ inline void wait_async0() {
  asm volatile("s_wait_asynccnt 0x0" ::: "memory");
}

// ---------------------------------------------------------------------------
// Prep kernels
// ---------------------------------------------------------------------------

__global__ void k_cvt_x(const float* __restrict__ x, u16* __restrict__ xbf, int n) {
  int i = blockIdx.x * blockDim.x + threadIdx.x;
  if (i < n) xbf[i] = f2bf(x[i]);
}

// Transpose+convert the 4 weight matrices: Wt[n*256+k] = W[k*256+n]
__global__ void k_cvt_w(const float* __restrict__ wq, const float* __restrict__ wk,
                        const float* __restrict__ wv, const float* __restrict__ wo,
                        u16* __restrict__ wt) {
  int i = blockIdx.x * blockDim.x + threadIdx.x;
  if (i >= D_ * D_) return;
  int n = i >> 8, k = i & 255;
  int src = k * D_ + n;
  wt[0 * D_ * D_ + i] = f2bf(wq[src]);
  wt[1 * D_ * D_ + i] = f2bf(wk[src]);
  wt[2 * D_ * D_ + i] = f2bf(wv[src]);
  wt[3 * D_ * D_ + i] = f2bf(wo[src]);
}

// Band softmax: bw[h][s], bwsqrt[h][s] = sqrt(bw + 1e-8)
__global__ void k_band(const float* __restrict__ logits,
                       float* __restrict__ bw, float* __restrict__ bws) {
  int h = threadIdx.x;
  if (h >= H_) return;
  float v[S_]; float mx = -1e30f;
  #pragma unroll
  for (int s = 0; s < S_; ++s) { v[s] = logits[h * S_ + s]; mx = fmaxf(mx, v[s]); }
  float sum = 0.f;
  #pragma unroll
  for (int s = 0; s < S_; ++s) { v[s] = __expf(v[s] - mx); sum += v[s]; }
  float inv = 1.0f / sum;
  #pragma unroll
  for (int s = 0; s < S_; ++s) {
    float w = v[s] * inv;
    bw[h * S_ + s]  = w;
    bws[h * S_ + s] = __builtin_sqrtf(w + 1e-8f);
  }
}

// Phase features Fpad[b,h,l,0..31] (cols 0..7 cos*bwsqrt, 8..15 sin*bwsqrt,
// 16..31 zero pad for the K=32 WMMA) and mag_head mg[b,h,l].
__global__ void k_feat(const float* __restrict__ cosp, const float* __restrict__ sinp,
                       const float* __restrict__ mag,  const float* __restrict__ bw,
                       const float* __restrict__ bws,  u16* __restrict__ fpad,
                       float* __restrict__ mg) {
  int idx = blockIdx.x * blockDim.x + threadIdx.x;   // B*H*L
  if (idx >= BH_ * L_) return;
  int l  = idx & (L_ - 1);
  int bh = idx >> 10;
  int h  = bh & (H_ - 1);
  int b  = bh >> 3;
  const float* cb = cosp + b * S_ * L_;  // [S][L]
  const float* sb = sinp + b * S_ * L_;
  const float* mb = mag  + b * S_ * L_;
  u16* fr = fpad + (size_t)idx * 32;
  float acc = 0.f;
  #pragma unroll
  for (int s = 0; s < S_; ++s) {
    float ws = bws[h * S_ + s];
    fr[s]      = f2bf(cb[s * L_ + l] * ws);
    fr[8 + s]  = f2bf(sb[s * L_ + l] * ws);
    acc       += mb[s * L_ + l] * bw[h * S_ + s];
  }
  #pragma unroll
  for (int s = 16; s < 32; ++s) fr[s] = 0;
  mg[idx] = acc;
}

// ---------------------------------------------------------------------------
// QKV projection: block = 4 waves covers one 16-row strip across all 256
// cols for one of {q,k,v}. Each wave: 1 A-fragment load reused for 4 col
// tiles (4 accumulators), K-loop of 8.
// ---------------------------------------------------------------------------
__global__ __launch_bounds__(128) void k_qkv(const u16* __restrict__ xbf,
                                             const u16* __restrict__ wt,
                                             const float* __restrict__ bq,
                                             const float* __restrict__ bk,
                                             const float* __restrict__ bv,
                                             u16* __restrict__ qbf,
                                             u16* __restrict__ kbf,
                                             u16* __restrict__ vt) {
  int rt = blockIdx.x;            // row tile 0..255
  int z  = blockIdx.y;            // 0=q 1=k 2=v
  int wave = threadIdx.x >> 5, lane = threadIdx.x & 31;
  const u16* W = wt + (size_t)z * D_ * D_;
  const float* bias = (z == 0) ? bq : (z == 1) ? bk : bv;
  int i0 = rt * 16;
  int n0w = wave * 64;            // this wave: cols n0w .. n0w+63 (4 tiles)

  v8f acc[4] = {{}, {}, {}, {}};
  #pragma unroll
  for (int kk = 0; kk < 8; ++kk) {
    v16bf a = load_a(xbf + (size_t)i0 * D_ + kk * 32, D_);
    #pragma unroll
    for (int c = 0; c < 4; ++c) {
      v16bf b = load_bt(W + (size_t)(n0w + c * 16) * D_ + kk * 32, D_);
      acc[c] = wmma_bf16(a, b, acc[c]);
    }
  }

  int m = lane & 15, kh = lane >> 4;
  #pragma unroll
  for (int c = 0; c < 4; ++c) {
    int nC = n0w + c * 16 + m;
    float bias_n = bias[nC];
    int h = nC >> 5, dh = nC & 31;
    #pragma unroll
    for (int r = 0; r < 8; ++r) {
      int g = i0 + r + (kh << 3);
      int b_ = g >> 10, l = g & (L_ - 1);
      u16 val = f2bf(acc[c][r] + bias_n);
      int bh = b_ * H_ + h;
      if (z == 0)      qbf[((size_t)bh * L_ + l) * DH_ + dh] = val;
      else if (z == 1) kbf[((size_t)bh * L_ + l) * DH_ + dh] = val;
      else             vt [((size_t)bh * DH_ + dh) * L_ + l] = val;
    }
  }
}

// ---------------------------------------------------------------------------
// Fused attention per (b,h): flash-style online softmax.
// Block = 128 threads (4 waves). Each wave owns TWO 16-row i-tiles (32 rows),
// j in tiles of 32. The 32-j K/F/V tiles (2KB each) are staged ONCE per block
// into LDS with async global->LDS copies (double-buffered, overlapped with
// compute); all 4 waves read B-fragments from LDS. P occupies a full K=32
// A-fragment (no zero padding waste in the PV WMMAs).
// ---------------------------------------------------------------------------
__global__ __launch_bounds__(128) void k_attn(const u16* __restrict__ qbf,
                                              const u16* __restrict__ kbf,
                                              const u16* __restrict__ vt,
                                              const u16* __restrict__ fpad,
                                              const float* __restrict__ mg,
                                              const float* __restrict__ mask,
                                              const float* __restrict__ pbs,
                                              u16* __restrict__ ctx) {
  __shared__ u16 kbuf[2][32 * 32];   // [j 0..31][d 0..31]
  __shared__ u16 fbuf[2][32 * 32];   // [j 0..31][f 0..31]
  __shared__ u16 vbuf[2][32 * 32];   // [d 0..31][j 0..31]
  __shared__ u16 lds_p[4][16 * 32];  // per-wave P transpose tile
  int tid = threadIdx.x;
  int wave = tid >> 5, lane = tid & 31;
  int m = lane & 15, kh = lane >> 4;
  int bh = blockIdx.y;
  int b = bh >> 3, h = bh & (H_ - 1);
  int i0 = blockIdx.x * 128 + wave * 32;   // two i-tiles: i0, i0+16

  const u16* Q = qbf + (size_t)bh * L_ * DH_;
  const u16* K = kbf + (size_t)bh * L_ * DH_;
  const u16* V = vt  + (size_t)bh * DH_ * L_;
  const u16* F = fpad + (size_t)bh * L_ * 32;
  const float* mgbh  = mg + (size_t)bh * L_;
  const float* maskb = mask + (size_t)b * L_;
  float pscale = __expf(pbs[h]);
  const float inv_sqrt_dh = 0.17677669529663687f;   // 1/sqrt(32)

  // Per-thread slice of the cooperative tile copy: 128 threads x 16B = 2KB.
  int srow = tid >> 2;           // 0..31
  int schk = (tid & 3) * 8;      // 0,8,16,24 (elements)
  auto stage = [&](int buf, int j0) {
    async_cp16((u32)(size_t)&kbuf[buf][srow * 32 + schk],
               K + (size_t)(j0 + srow) * DH_ + schk);
    async_cp16((u32)(size_t)&fbuf[buf][srow * 32 + schk],
               F + (size_t)(j0 + srow) * 32 + schk);
    async_cp16((u32)(size_t)&vbuf[buf][srow * 32 + schk],
               V + (size_t)srow * L_ + j0 + schk);
  };

  stage(0, 0);   // prologue: stage first tile (overlaps with setup below)

  v16bf aq[2], af[2];
  float miv[2][8], rmx[2][8], rsm[2][8];
  v8f o[2][2];
  #pragma unroll
  for (int t = 0; t < 2; ++t) {
    int it = i0 + t * 16;
    aq[t] = load_a(Q + (size_t)it * DH_, DH_);
    af[t] = load_a(F + (size_t)it * 32, 32);
    o[t][0] = (v8f){}; o[t][1] = (v8f){};
    #pragma unroll
    for (int r = 0; r < 8; ++r) {
      miv[t][r] = mgbh[it + r + (kh << 3)];
      rmx[t][r] = -1e30f;
      rsm[t][r] = 0.f;
    }
  }

  wait_async0();
  __syncthreads();

  u16* pt = lds_p[wave];
  int cb = 0;

  for (int jt = 0; jt < L_ / 32; ++jt) {
    int j0 = jt * 32;
    if (jt + 1 < L_ / 32) stage(cb ^ 1, j0 + 32);  // overlap copy with compute

    // B-fragments from LDS (shared by all 4 waves, reused by both i-tiles).
    v16bf bk0 = load_bt(&kbuf[cb][0], 32);
    v16bf bk1 = load_bt(&kbuf[cb][16 * 32], 32);
    v16bf bf0 = load_bt(&fbuf[cb][0], 32);
    v16bf bf1 = load_bt(&fbuf[cb][16 * 32], 32);
    Frag16 fv0, fv1;
    {
      const u16* vr0 = &vbuf[cb][m * 32 + (kh << 4)];
      const u16* vr1 = &vbuf[cb][(16 + m) * 32 + (kh << 4)];
      fv0.q[0] = *(const uint4*)(vr0); fv0.q[1] = *(const uint4*)(vr0 + 8);
      fv1.q[0] = *(const uint4*)(vr1); fv1.q[1] = *(const uint4*)(vr1 + 8);
    }

    float mj0 = mgbh[j0 + m],  mj1 = mgbh[j0 + 16 + m];
    float mk0 = maskb[j0 + m], mk1 = maskb[j0 + 16 + m];

    #pragma unroll
    for (int t = 0; t < 2; ++t) {
      v8f zero = {};
      v8f s0 = wmma_bf16(aq[t], bk0, zero);
      v8f s1 = wmma_bf16(aq[t], bk1, zero);
      v8f p0 = wmma_bf16(af[t], bf0, zero);
      v8f p1 = wmma_bf16(af[t], bf1, zero);

      float pe0[8], pe1[8];
      #pragma unroll
      for (int r = 0; r < 8; ++r) {
        float gi = miv[t][r];
        float sc0 = s0[r] * inv_sqrt_dh
                  + p0[r] * __builtin_sqrtf(gi * mj0 + 1e-8f) * pscale + mk0;
        float sc1 = s1[r] * inv_sqrt_dh
                  + p1[r] * __builtin_sqrtf(gi * mj1 + 1e-8f) * pscale + mk1;
        float rm = rmax16(fmaxf(sc0, sc1));
        float nm = fmaxf(rmx[t][r], rm);
        float alpha = __expf(rmx[t][r] - nm);
        float e0 = __expf(sc0 - nm);
        float e1 = __expf(sc1 - nm);
        float ps = rsum16(e0 + e1);
        rsm[t][r] = rsm[t][r] * alpha + ps;
        rmx[t][r] = nm;
        o[t][0][r] *= alpha;
        o[t][1][r] *= alpha;
        pe0[r] = e0; pe1[r] = e1;
      }

      // Transpose P (C-layout -> A-layout) through this wave's LDS tile.
      #pragma unroll
      for (int r = 0; r < 8; ++r) {
        int M = r + (kh << 3);
        pt[M * 32 + m]      = f2bf(pe0[r]);
        pt[M * 32 + 16 + m] = f2bf(pe1[r]);
      }
      asm volatile("s_wait_dscnt 0" ::: "memory");  // wave-lockstep: no barrier

      Frag16 ap;
      ap.q[0] = *(const uint4*)(pt + m * 32 + kh * 16);
      ap.q[1] = *(const uint4*)(pt + m * 32 + kh * 16 + 8);

      o[t][0] = wmma_bf16(ap.bf, fv0.bf, o[t][0]);  // context cols 0..15
      o[t][1] = wmma_bf16(ap.bf, fv1.bf, o[t][1]);  // context cols 16..31
    }

    wait_async0();      // next tile's copies complete
    __syncthreads();    // all waves done reading buf[cb]; copies visible
    cb ^= 1;
  }

  #pragma unroll
  for (int t = 0; t < 2; ++t) {
    #pragma unroll
    for (int r = 0; r < 8; ++r) {
      float inv = 1.0f / rsm[t][r];
      int l = i0 + t * 16 + r + (kh << 3);
      u16* crow = ctx + ((size_t)(b * L_ + l)) * D_ + h * DH_;
      crow[m]      = f2bf(o[t][0][r] * inv);
      crow[16 + m] = f2bf(o[t][1][r] * inv);
    }
  }
}

// ---------------------------------------------------------------------------
// Output projection + bias + residual + LayerNorm.
// Block = 256 threads (8 waves) per 16-row tile; each wave does 2 col tiles.
// ---------------------------------------------------------------------------
__global__ __launch_bounds__(256) void k_out_ln(const u16* __restrict__ ctx,
                                                const u16* __restrict__ wot,
                                                const float* __restrict__ bo,
                                                const float* __restrict__ resid,
                                                const float* __restrict__ gamma,
                                                const float* __restrict__ beta,
                                                float* __restrict__ out) {
  __shared__ float xls[16][D_ + 4];
  int wave = threadIdx.x >> 5, lane = threadIdx.x & 31;
  int m = lane & 15, kh = lane >> 4;
  int i0 = blockIdx.x * 16;
  int n0 = wave * 32;   // this wave: cols n0..n0+31 (two 16-col tiles)

  v8f acc0 = {}, acc1 = {};
  #pragma unroll
  for (int kk = 0; kk < 8; ++kk) {
    v16bf a  = load_a(ctx + (size_t)i0 * D_ + kk * 32, D_);
    v16bf b0 = load_bt(wot + (size_t)n0 * D_ + kk * 32, D_);
    v16bf b1 = load_bt(wot + (size_t)(n0 + 16) * D_ + kk * 32, D_);
    acc0 = wmma_bf16(a, b0, acc0);
    acc1 = wmma_bf16(a, b1, acc1);
  }

  int c0 = n0 + m, c1 = n0 + 16 + m;
  float bo0 = bo[c0], bo1 = bo[c1];
  #pragma unroll
  for (int r = 0; r < 8; ++r) {
    int M = r + (kh << 3);
    int g = i0 + M;
    xls[M][c0] = acc0[r] + bo0 + resid[(size_t)g * D_ + c0];
    xls[M][c1] = acc1[r] + bo1 + resid[(size_t)g * D_ + c1];
  }
  __syncthreads();

  // LayerNorm: 16 threads per row, 16 cols each; shfl reduction in 16-groups.
  int row = threadIdx.x >> 4, cg = threadIdx.x & 15;
  float sum = 0.f, sq = 0.f, xv[16];
  #pragma unroll
  for (int c = 0; c < 16; ++c) {
    float x = xls[row][cg * 16 + c];
    xv[c] = x; sum += x; sq += x * x;
  }
  sum = rsum16(sum);
  sq  = rsum16(sq);
  float mu   = sum * (1.0f / D_);
  float var  = sq * (1.0f / D_) - mu * mu;
  float rstd = rsqrtf(fmaxf(var, 0.f) + 1e-12f);
  int g = i0 + row;
  #pragma unroll
  for (int c = 0; c < 16; ++c) {
    int col = cg * 16 + c;
    out[(size_t)g * D_ + col] = (xv[c] - mu) * rstd * gamma[col] + beta[col];
  }
}

// ---------------------------------------------------------------------------
// Launcher
// ---------------------------------------------------------------------------
extern "C" void kernel_launch(void* const* d_in, const int* in_sizes, int n_in,
                              void* d_out, int out_size, void* d_ws, size_t ws_size,
                              hipStream_t stream) {
  (void)in_sizes; (void)n_in; (void)out_size; (void)ws_size;
  const float* hidden = (const float*)d_in[0];
  const float* amask  = (const float*)d_in[1];
  const float* cosp   = (const float*)d_in[2];
  const float* sinp   = (const float*)d_in[3];
  const float* mag    = (const float*)d_in[4];
  const float* Wq = (const float*)d_in[5];  const float* bq = (const float*)d_in[6];
  const float* Wk = (const float*)d_in[7];  const float* bk = (const float*)d_in[8];
  const float* Wv = (const float*)d_in[9];  const float* bv = (const float*)d_in[10];
  const float* Wo = (const float*)d_in[11]; const float* bo = (const float*)d_in[12];
  const float* blog = (const float*)d_in[13];
  const float* pbs  = (const float*)d_in[14];
  const float* lng  = (const float*)d_in[15];
  const float* lnb  = (const float*)d_in[16];
  float* out = (float*)d_out;

  // Workspace layout (bytes)
  char* w = (char*)d_ws;
  u16*   xbf  = (u16*)(w + 0);                   // 4096*256 bf16   (2 MB)
  u16*   wt   = (u16*)(w + 2097152);             // 4x 256*256 bf16 (512 KB)
  u16*   qbf  = (u16*)(w + 2621440);             // [B,H,L,DH] bf16 (2 MB)
  u16*   kbf  = (u16*)(w + 4718592);             // [B,H,L,DH] bf16 (2 MB)
  u16*   vt   = (u16*)(w + 6815744);             // [B,H,DH,L] bf16 (2 MB)
  u16*   fpad = (u16*)(w + 8912896);             // [B,H,L,32] bf16 (2 MB)
  float* mg   = (float*)(w + 11010048);          // [B,H,L] f32     (128 KB)
  u16*   ctx  = (u16*)(w + 11141120);            // [B,L,D] bf16    (2 MB)
  float* bw   = (float*)(w + 13238272);          // [H,S] f32
  float* bws  = (float*)(w + 13238528);          // [H,S] f32

  // Prep
  k_cvt_x<<<(BL_ * D_ + 255) / 256, 256, 0, stream>>>(hidden, xbf, BL_ * D_);
  k_cvt_w<<<(D_ * D_ + 255) / 256, 256, 0, stream>>>(Wq, Wk, Wv, Wo, wt);
  k_band<<<1, 32, 0, stream>>>(blog, bw, bws);
  k_feat<<<(BH_ * L_ + 255) / 256, 256, 0, stream>>>(cosp, sinp, mag, bw, bws, fpad, mg);

  // QKV projection (4 col tiles per wave, A-fragment reuse)
  k_qkv<<<dim3(BL_ / 16, 3), 128, 0, stream>>>(
      xbf, wt, bq, bk, bv, qbf, kbf, vt);

  // Fused flash attention with phase-sync bias
  // (async LDS staging, 2 i-tiles/wave, j-tile=32)
  k_attn<<<dim3(L_ / 128, BH_), 128, 0, stream>>>(
      qbf, kbf, vt, fpad, mg, amask, pbs, ctx);

  // Output projection + residual + LayerNorm
  k_out_ln<<<BL_ / 16, 256, 0, stream>>>(
      ctx, wt + (size_t)3 * D_ * D_, bo, hidden, lng, lnb, out);
}
